// NeuralCDE_18743237280219
// MI455X (gfx1250) — matrix operand order repaired and checked by reading, not compiled, not run on hardware
//
#include <hip/hip_runtime.h>
#include <hip/hip_bf16.h>

// Neural CDE persistent-scan kernel for gfx1250 (MI455X).
// grid = 8 workgroups x 256 threads (8 wave32). Each WG owns 16 batch rows and
// runs the full 2000-step Euler scan with all state resident in LDS/VGPRs.
// GEMMs use v_wmma_f32_16x16x32_f16 (f16 in, f32 accumulate); weight B-fragments
// are loaded once and pinned in VGPRs for the whole loop.

#define B_TOT     128
#define NP        128
#define D_CH      8
#define H_DIM     32
#define WID       128
#define NC        10
#define DT0       0.01f
#define NUM_STEPS 2000

#define ROWS      16    // batch rows per workgroup
#define NTHREADS  256   // 8 waves (wave32)

#define SP        136   // f16 activation row stride (halves), padded vs 128
#define FS        264   // f32 tanh-output row stride (floats), padded vs 256
#define YS        36    // f32 y row stride, padded vs 32

typedef __attribute__((ext_vector_type(16))) _Float16 v16h;
typedef __attribute__((ext_vector_type(8)))  float    v8f;

__device__ __forceinline__ float sp_f(float x) {
  // softplus, overflow-safe: max(x,0) + log1p(exp(-|x|))
  return fmaxf(x, 0.0f) + __logf(1.0f + __expf(-fabsf(x)));
}
__device__ __forceinline__ float tanh_f(float x) {
  float xc = fminf(fmaxf(x, -15.0f), 15.0f);
  float e2 = __expf(2.0f * xc);
  return (e2 - 1.0f) / (e2 + 1.0f);
}

// B-operand fragment from global row-major weight W (Ntot x Ktot, f32).
// Tile (nt, kt): lane holds column n = nt*16 + lane%16; j=0..7 -> kb+j,
// j=8..15 -> kb+16+(j-8), kb = kt*32 + (lane/16)*8 (matches A layout).
__device__ __forceinline__ v16h load_bfrag(const float* __restrict__ W, int Ktot,
                                           int nt, int kt, int lane) {
  const int n  = nt * 16 + (lane & 15);
  const int kb = kt * 32 + ((lane >> 4) << 3);
  const float* row = W + (size_t)n * Ktot + kb;
  v16h b;
#pragma unroll
  for (int j = 0; j < 8; ++j) { b[j] = (_Float16)row[j]; b[8 + j] = (_Float16)row[16 + j]; }
  return b;
}

// A-operand fragment from f16 row-major LDS buffer (stride SP halves).
// Lane holds row m = lane%16; A layout interleaves K: j=0..7 -> kb+j,
// j=8..15 -> kb+16+(j-8), kb = kt*32 + (lane/16)*8. Two 16B contiguous chunks.
__device__ __forceinline__ v16h load_afrag(const _Float16* buf, int kt, int lane) {
  const int m  = lane & 15;
  const int kb = kt * 32 + ((lane >> 4) << 3);
  const _Float16* p = buf + m * SP + kb;
  v16h a;
#pragma unroll
  for (int j = 0; j < 8; ++j) { a[j] = p[j]; a[8 + j] = p[16 + j]; }
  return a;
}

__global__ __launch_bounds__(NTHREADS, 1)
void ncde_scan_kernel(const float* __restrict__ ts,
                      const float* __restrict__ cd, const float* __restrict__ cc,
                      const float* __restrict__ cb, const float* __restrict__ ca,
                      const float* __restrict__ W0, const float* __restrict__ b0,
                      const float* __restrict__ W1, const float* __restrict__ b1,
                      const float* __restrict__ W2, const float* __restrict__ b2,
                      const float* __restrict__ F0, const float* __restrict__ f0,
                      const float* __restrict__ F1, const float* __restrict__ f1,
                      const float* __restrict__ F2, const float* __restrict__ f2,
                      const float* __restrict__ R,  const float* __restrict__ rb,
                      float* __restrict__ out)
{
  __shared__ float    s_ts[NP];
  __shared__ float    s_y[ROWS * YS];       // fp32 hidden state, padded
  __shared__ _Float16 s_act[ROWS * SP];     // f16 inter-GEMM activations, padded
  __shared__ float    s_f[ROWS * FS];       // fp32 tanh output / init-MLP scratch
  __shared__ float    s_dx[ROWS * D_CH];    // per-step dX

  const int tid  = threadIdx.x;
  const int w    = tid >> 5;                // wave id 0..7
  const int lane = tid & 31;
  const int bg0  = blockIdx.x * ROWS;       // first global batch row

  for (int i = tid; i < NP; i += NTHREADS) s_ts[i] = ts[i];

  // ---- Pin all weight B-fragments (f16) in VGPRs: 13 fragments/wave ----
  v16h bF0 = load_bfrag(F0, H_DIM, w, 0, lane);               // GEMM1: K=32
  v16h bF1[4], bF2a[4], bF2b[4];
#pragma unroll
  for (int kt = 0; kt < 4; ++kt) {
    bF1[kt]  = load_bfrag(F1, WID, w,         kt, lane);      // GEMM2
    bF2a[kt] = load_bfrag(F2, WID, 2 * w + 0, kt, lane);      // GEMM3 tile A
    bF2b[kt] = load_bfrag(F2, WID, 2 * w + 1, kt, lane);      // GEMM3 tile B
  }
  const float bias1  = f0[w * 16 + (lane & 15)];
  const float bias2  = f1[w * 16 + (lane & 15)];
  const float bias3a = f2[(2 * w + 0) * 16 + (lane & 15)];
  const float bias3b = f2[(2 * w + 1) * 16 + (lane & 15)];

  // ---- Initial MLP (runs once; VALU is fine) ----
  if (tid < ROWS * D_CH) {              // x = coeff_a[:, 0, :]  (reuse s_dx)
    int b = tid >> 3, d = tid & 7;
    s_dx[tid] = ca[(size_t)(bg0 + b) * (NP - 1) * D_CH + d];
  }
  __syncthreads();
  for (int o = tid; o < ROWS * WID; o += NTHREADS) {   // h0 -> s_f[:,0:128)
    int b = o >> 7, n = o & 127;
    float s = b0[n];
#pragma unroll
    for (int k = 0; k < D_CH; ++k) s += s_dx[b * D_CH + k] * W0[n * D_CH + k];
    s_f[b * FS + n] = sp_f(s);
  }
  __syncthreads();
  for (int o = tid; o < ROWS * WID; o += NTHREADS) {   // h1 -> s_f[:,128:256)
    int b = o >> 7, n = o & 127;
    float s = b1[n];
    for (int k = 0; k < WID; ++k) s += s_f[b * FS + k] * W1[n * WID + k];
    s_f[b * FS + WID + n] = sp_f(s);
  }
  __syncthreads();
  for (int o = tid; o < ROWS * H_DIM; o += NTHREADS) { // y0
    int b = o >> 5, h = o & 31;
    float s = b2[h];
    for (int k = 0; k < WID; ++k) s += s_f[b * FS + WID + k] * W2[h * WID + k];
    s_y[b * YS + h] = s;
  }
  __syncthreads();

  const float ts0 = s_ts[0];
  const float tsL = s_ts[NP - 1];
  const float inv_sp = (float)(NP - 1) / (tsL - ts0);
  const int mb   = (lane >> 4) << 3;        // C/D: row base per half-wave
  const int ncol = w * 16 + (lane & 15);    // C/D: column this lane owns

  // =================== 2000-step Euler scan ===================
  for (int n = 0; n < NUM_STEPS; ++n) {
    const float t = ts0 + DT0 * (float)n;
    int idx = (int)((t - ts0) * inv_sp);
    idx = idx < 0 ? 0 : (idx > NP - 2 ? NP - 2 : idx);
    if (idx < NP - 2 && s_ts[idx + 1] <= t) ++idx;     // fp-rounding fixups
    if (idx > 0 && s_ts[idx] > t) --idx;
    const float fr  = t - s_ts[idx];
    const float dtn = fminf(DT0, tsL - t);

    if (tid < ROWS * D_CH) {                           // dX[b,d] spline eval
      int b = tid >> 3, d = tid & 7;
      size_t gi = ((size_t)(bg0 + b) * (NP - 1) + idx) * D_CH + d;
      s_dx[tid] = cb[gi] + 2.0f * cc[gi] * fr + 3.0f * cd[gi] * fr * fr;
    }

    // GEMM1: softplus(y @ F0^T + f0); K = H = 32 -> one WMMA per wave
    {
      const int m  = lane & 15;
      const int kb = (lane >> 4) << 3;
      v16h a;
#pragma unroll
      for (int j = 0; j < 8; ++j) {
        a[j]     = (_Float16)s_y[m * YS + kb + j];
        a[8 + j] = (_Float16)s_y[m * YS + kb + 16 + j];
      }
      v8f c = {};
      c = __builtin_amdgcn_wmma_f32_16x16x32_f16(false, a, false, bF0, (short)0, c, false, false);
#pragma unroll
      for (int r = 0; r < 8; ++r)
        s_act[(r + mb) * SP + ncol] = (_Float16)sp_f(c[r] + bias1);
    }
    __syncthreads();

    // GEMM2: softplus(h @ F1^T + f1); K = 128 -> 4 chained WMMAs
    v16h a0 = load_afrag(s_act, 0, lane);
    v16h a1 = load_afrag(s_act, 1, lane);
    v16h a2 = load_afrag(s_act, 2, lane);
    v16h a3 = load_afrag(s_act, 3, lane);
    __syncthreads();                                   // reads done before overwrite
    {
      v8f c = {};
      c = __builtin_amdgcn_wmma_f32_16x16x32_f16(false, a0, false, bF1[0], (short)0, c, false, false);
      c = __builtin_amdgcn_wmma_f32_16x16x32_f16(false, a1, false, bF1[1], (short)0, c, false, false);
      c = __builtin_amdgcn_wmma_f32_16x16x32_f16(false, a2, false, bF1[2], (short)0, c, false, false);
      c = __builtin_amdgcn_wmma_f32_16x16x32_f16(false, a3, false, bF1[3], (short)0, c, false, false);
#pragma unroll
      for (int r = 0; r < 8; ++r)
        s_act[(r + mb) * SP + ncol] = (_Float16)sp_f(c[r] + bias2);
    }
    __syncthreads();

    // GEMM3: tanh(h @ F2^T + f2); 2 N-tiles x 4 K-steps per wave
    a0 = load_afrag(s_act, 0, lane);
    a1 = load_afrag(s_act, 1, lane);
    a2 = load_afrag(s_act, 2, lane);
    a3 = load_afrag(s_act, 3, lane);
    {
      v8f c = {};
      c = __builtin_amdgcn_wmma_f32_16x16x32_f16(false, a0, false, bF2a[0], (short)0, c, false, false);
      c = __builtin_amdgcn_wmma_f32_16x16x32_f16(false, a1, false, bF2a[1], (short)0, c, false, false);
      c = __builtin_amdgcn_wmma_f32_16x16x32_f16(false, a2, false, bF2a[2], (short)0, c, false, false);
      c = __builtin_amdgcn_wmma_f32_16x16x32_f16(false, a3, false, bF2a[3], (short)0, c, false, false);
      const int nc0 = (2 * w + 0) * 16 + (lane & 15);
#pragma unroll
      for (int r = 0; r < 8; ++r)
        s_f[(r + mb) * FS + nc0] = tanh_f(c[r] + bias3a);

      v8f c2 = {};
      c2 = __builtin_amdgcn_wmma_f32_16x16x32_f16(false, a0, false, bF2b[0], (short)0, c2, false, false);
      c2 = __builtin_amdgcn_wmma_f32_16x16x32_f16(false, a1, false, bF2b[1], (short)0, c2, false, false);
      c2 = __builtin_amdgcn_wmma_f32_16x16x32_f16(false, a2, false, bF2b[2], (short)0, c2, false, false);
      c2 = __builtin_amdgcn_wmma_f32_16x16x32_f16(false, a3, false, bF2b[3], (short)0, c2, false, false);
      const int nc1 = (2 * w + 1) * 16 + (lane & 15);
#pragma unroll
      for (int r = 0; r < 8; ++r)
        s_f[(r + mb) * FS + nc1] = tanh_f(c2[r] + bias3b);
    }
    __syncthreads();

    // y[b,h] += dt * sum_d f[b,h,d] * dX[b,d]   (512 values, 2 per thread)
#pragma unroll
    for (int rep = 0; rep < 2; ++rep) {
      int o = tid + rep * NTHREADS;
      int b = o >> 5, h = o & 31;
      float s = 0.0f;
#pragma unroll
      for (int d = 0; d < D_CH; ++d)
        s += s_f[b * FS + h * D_CH + d] * s_dx[b * D_CH + d];
      s_y[b * YS + h] += dtn * s;
    }
    __syncthreads();
  }

  // ---- logits = y @ R^T + rb ----
  for (int o = tid; o < ROWS * NC; o += NTHREADS) {
    int b = o / NC, c = o % NC;
    float s = rb[c];
#pragma unroll
    for (int h = 0; h < H_DIM; ++h) s += s_y[b * YS + h] * R[c * H_DIM + h];
    out[(size_t)(bg0 + b) * NC + c] = s;
  }
}

extern "C" void kernel_launch(void* const* d_in, const int* in_sizes, int n_in,
                              void* d_out, int out_size, void* d_ws, size_t ws_size,
                              hipStream_t stream) {
  (void)in_sizes; (void)n_in; (void)out_size; (void)d_ws; (void)ws_size;
  const float* ts = (const float*)d_in[0];
  const float* cd = (const float*)d_in[1];
  const float* cc = (const float*)d_in[2];
  const float* cb = (const float*)d_in[3];
  const float* ca = (const float*)d_in[4];
  const float* W0 = (const float*)d_in[5];
  const float* b0 = (const float*)d_in[6];
  const float* W1 = (const float*)d_in[7];
  const float* b1 = (const float*)d_in[8];
  const float* W2 = (const float*)d_in[9];
  const float* b2 = (const float*)d_in[10];
  const float* F0 = (const float*)d_in[11];
  const float* f0 = (const float*)d_in[12];
  const float* F1 = (const float*)d_in[13];
  const float* f1 = (const float*)d_in[14];
  const float* F2 = (const float*)d_in[15];
  const float* f2 = (const float*)d_in[16];
  const float* R  = (const float*)d_in[17];
  const float* rb = (const float*)d_in[18];
  ncde_scan_kernel<<<dim3(B_TOT / ROWS), dim3(NTHREADS), 0, stream>>>(
      ts, cd, cc, cb, ca, W0, b0, W1, b1, W2, b2,
      F0, f0, F1, f1, F2, f2, R, rb, (float*)d_out);
}
